// EHRMemoryAttention_41875931136791
// MI455X (gfx1250) — compile-verified, matrix operand order checked by reading, not run on hardware
//
#include <hip/hip_runtime.h>
#include <stdint.h>

// ---------------- problem constants ----------------
#define DMODEL 256
#define NHEAD  8
#define DHEAD  32
#define TOPN   32
#define NQTOT  16384
#define MMEM   2048
#define EPSLN  1e-5f
#define NEGSL  0.01f

// ---------------- CDNA5 WMMA types ----------------
typedef __attribute__((ext_vector_type(16))) __bf16 v16bf;
typedef __attribute__((ext_vector_type(8)))  float  v8f;

__device__ __forceinline__ v8f wmma_bf16(v16bf a, v16bf b, v8f c) {
  // D = A(16x32 bf16) * B(32x16 bf16) + C(16x16 f32)
  return __builtin_amdgcn_wmma_f32_16x16x32_bf16(false, a, false, b,
                                                 (short)0, c, false, false);
}

// round-to-nearest-even f32 -> bf16 (bit pattern as ushort)
__device__ __forceinline__ unsigned short f2bf(float f) {
  unsigned u = __float_as_uint(f);
  unsigned r = u + 0x7FFFu + ((u >> 16) & 1u);
  return (unsigned short)(r >> 16);
}

// inverse of the monotonic float->uint order-preserving map
__device__ __forceinline__ float keyf(unsigned k) {
  unsigned u = (k & 0x80000000u) ? (k & 0x7FFFFFFFu) : ~k;
  return __uint_as_float(u);
}

// Load a 16-value bf16 fragment for one matrix row (32-elem K span),
// following the CDNA5 16-bit A/B VGPR layout:
//   lanes 0-15  hold K = {0..7, 16..23}
//   lanes 16-31 hold K = {8..15, 24..31}
__device__ __forceinline__ v16bf load_frag(const unsigned short* rowbase, int hiHalf) {
  union { v16bf v; uint4 q[2]; } u;
  const unsigned short* p = rowbase + (hiHalf ? 8 : 0);
  u.q[0] = *(const uint4*)p;
  u.q[1] = *(const uint4*)(p + 16);
  return u.v;
}

// ---------------- kernel 0: f32 -> bf16 convert ----------------
__global__ void cvt_bf16_kernel(const float* __restrict__ src,
                                unsigned short* __restrict__ dst, int n) {
  int i = blockIdx.x * blockDim.x + threadIdx.x;
  int stride = gridDim.x * blockDim.x;
  for (; i < n; i += stride) dst[i] = f2bf(src[i]);
}

// ---------------- kernel 1: Y = X @ W^T + b  (bf16 in/out) ----------------
// block = 16 output rows, 256 threads (8 waves); each wave does 2 n-tiles x 8 k-WMMAs.
// transposed!=0 stores Y^T with leading dim ldT (used for V -> Vt[D][M]).
__global__ void __launch_bounds__(256)
proj_gemm_kernel(const unsigned short* __restrict__ Xb,
                 const unsigned short* __restrict__ Wb,
                 const float* __restrict__ bias,
                 unsigned short* __restrict__ Y,
                 int transposed, int ldT) {
  const int tid  = threadIdx.x;
  const int wave = tid >> 5, lane = tid & 31;
  const int hiL  = lane >> 4, l16 = lane & 15;
  const int rowbase = blockIdx.x * 16;
  const unsigned short* xrow = Xb + (size_t)(rowbase + l16) * DMODEL;

#pragma unroll
  for (int t = 0; t < 2; ++t) {
    const int n0 = (wave * 2 + t) * 16;
    const unsigned short* wrow = Wb + (size_t)(n0 + l16) * DMODEL;
    v8f c = {};
#pragma unroll
    for (int kt = 0; kt < 8; ++kt)
      c = wmma_bf16(load_frag(xrow + kt * 32, hiL),
                    load_frag(wrow + kt * 32, hiL), c);
    const float bv = bias[n0 + l16];
#pragma unroll
    for (int r = 0; r < 8; ++r) {
      const int m = r + 8 * hiL;           // output row within tile
      const int n = n0 + l16;              // output col
      const unsigned short o = f2bf(c[r] + bv);
      if (transposed) Y[(size_t)n * ldT + rowbase + m] = o;
      else            Y[(size_t)(rowbase + m) * DMODEL + n] = o;
    }
  }
}

// ---------------- kernel 2: fused top-32 attention ----------------
// grid = (NQ/16, H), 256 threads. Scores [16 x 2048] f32 live in LDS.
#define SSTR 2065                 // f32 score row stride (padded, bank-spread)
#define PSTR 2056                 // bf16 prob row stride (16B-aligned rows)
#define SMEM_P   (16 * SSTR * 4)
#define SMEM_CTX (SMEM_P + 16 * PSTR * 2)
#define SMEM_TOT (SMEM_CTX + 16 * 32 * 4)

__global__ void __launch_bounds__(256)
attn_kernel(const unsigned short* __restrict__ Qb,   // [NQ][256] bf16
            const unsigned short* __restrict__ Kb,   // [M][256]  bf16
            const unsigned short* __restrict__ Vt,   // [256][M]  bf16 (V^T)
            unsigned short* __restrict__ Ctx) {      // [NQ][256] bf16
  extern __shared__ __align__(16) unsigned char smem[];
  float*          scS  = (float*)(smem);
  unsigned short* pS   = (unsigned short*)(smem + SMEM_P);
  float*          ctxS = (float*)(smem + SMEM_CTX);

  const int tid  = threadIdx.x;
  const int wave = tid >> 5, lane = tid & 31;
  const int hiL  = lane >> 4, l16 = lane & 15;
  const int qbase = blockIdx.x * 16;
  const int h     = blockIdx.y;

  // ---- phase 0: zero P (softmax probs) and ctx accumulator ----
  {
    uint4 z = {};
    uint4* pz = (uint4*)pS;
    const int nv = (16 * PSTR * 2) / 16;
    for (int i = tid; i < nv; i += 256) pz[i] = z;
    for (int i = tid; i < 16 * 32; i += 256) ctxS[i] = 0.0f;
  }

  // ---- phase 1: scores = (Q tile) @ (K tile)^T / sqrt(DH) -> LDS ----
  const unsigned short* qrow = Qb + (size_t)(qbase + l16) * DMODEL + h * DHEAD;
  const v16bf aq = load_frag(qrow, hiL);
  const float scale = 0.17677669529663687f;   // 1/sqrt(32)
  for (int kt = wave; kt < MMEM / 16; kt += 8) {
    const unsigned short* krow = Kb + (size_t)(kt * 16 + l16) * DMODEL + h * DHEAD;
    __builtin_prefetch(Kb + (size_t)((kt + 8) * 16 + l16) * DMODEL + h * DHEAD, 0, 1);
    v8f c = {};
    c = wmma_bf16(aq, load_frag(krow, hiL), c);
#pragma unroll
    for (int r = 0; r < 8; ++r) {
      const int m = r + 8 * hiL;
      scS[m * SSTR + kt * 16 + l16] = c[r] * scale;
    }
  }
  __syncthreads();

  // ---- phase 2: per-row exact top-32 threshold (binary search in key space) ----
  // 16 lanes per row; each lane caches 128 scores in VGPRs.
  const int row = tid >> 4;
  const int j   = tid & 15;
  float sv[128];
#pragma unroll
  for (int cc = 0; cc < 128; ++cc) sv[cc] = scS[row * SSTR + j + (cc << 4)];

  unsigned lo = 0u, hiK = 0xFFFFFFF0u;
  for (int it = 0; it < 32; ++it) {
    const unsigned mid = lo + ((hiK - lo + 1u) >> 1);
    const float midf = keyf(mid);
    int cnt = 0;
#pragma unroll
    for (int cc = 0; cc < 128; ++cc) cnt += (sv[cc] >= midf) ? 1 : 0;
#pragma unroll
    for (int o = 8; o; o >>= 1) cnt += __shfl_xor(cnt, o, 32);  // 16-lane group sum
    if (cnt >= TOPN) lo = mid; else hiK = mid - 1u;
  }
  const float thr = keyf(lo);   // == 32nd-largest score; keep all >= thr (ties kept)

  // ---- phase 3: masked softmax -> bf16 probs in LDS ----
  float mx = -3.0e38f;
#pragma unroll
  for (int cc = 0; cc < 128; ++cc) if (sv[cc] >= thr) mx = fmaxf(mx, sv[cc]);
#pragma unroll
  for (int o = 8; o; o >>= 1) mx = fmaxf(mx, __shfl_xor(mx, o, 32));
  float se = 0.0f;
#pragma unroll
  for (int cc = 0; cc < 128; ++cc) if (sv[cc] >= thr) se += __expf(sv[cc] - mx);
#pragma unroll
  for (int o = 8; o; o >>= 1) se += __shfl_xor(se, o, 32);
  const float inv = 1.0f / se;
#pragma unroll
  for (int cc = 0; cc < 128; ++cc)
    if (sv[cc] >= thr)
      pS[row * PSTR + j + (cc << 4)] = f2bf(__expf(sv[cc] - mx) * inv);
  __syncthreads();

  // ---- phase 4: ctx(16x32) = P(16x2048) @ V(2048x32) via bf16 WMMA ----
  {
    const int n0 = (wave & 1) * 16;
    const unsigned short* vrow = Vt + (size_t)(h * DHEAD + n0 + l16) * MMEM;
    v8f c = {};
#pragma unroll
    for (int i = 0; i < 16; ++i) {
      const int k0 = (((wave >> 1) + (i << 2)) << 5);   // k-tiles 0..63, 4 waves/n-tile
      c = wmma_bf16(load_frag(pS + l16 * PSTR + k0, hiL),
                    load_frag(vrow + k0, hiL), c);
    }
#pragma unroll
    for (int r = 0; r < 8; ++r)
      atomicAdd(&ctxS[(r + 8 * hiL) * 32 + n0 + l16], c[r]);   // ds_add_f32
  }
  __syncthreads();

  // ---- phase 5: write ctx tile as bf16 pairs ----
  {
    const int e = tid * 2;
    const int m = e >> 5, col = e & 31;
    const unsigned short b0 = f2bf(ctxS[m * 32 + col]);
    const unsigned short b1 = f2bf(ctxS[m * 32 + col + 1]);
    const unsigned pack = (unsigned)b0 | ((unsigned)b1 << 16);
    *(unsigned*)&Ctx[(size_t)(qbase + m) * DMODEL + h * DHEAD + col] = pack;
  }
}

// ---------------- kernel 3: fused epilogue ----------------
// att_out = ctx@Wo^T+bo ; x1 = LN1(x+att_out) ; h = leaky(x1@W1^T+b1)
// ff = h@W2^T+b2 ; out = LN2(x1+ff).  One 16-row tile per block.
__global__ void __launch_bounds__(256)
epilogue_kernel(const unsigned short* __restrict__ Ctx, const float* __restrict__ X,
                const unsigned short* __restrict__ Wo, const float* __restrict__ bo,
                const float* __restrict__ g1, const float* __restrict__ be1,
                const unsigned short* __restrict__ W1, const float* __restrict__ b1,
                const unsigned short* __restrict__ W2, const float* __restrict__ b2,
                const float* __restrict__ g2, const float* __restrict__ be2,
                float* __restrict__ Out) {
  __shared__ float          accS[16][260];
  __shared__ unsigned short stageS[16][264];
  __shared__ float          x1S[16][260];

  const int tid  = threadIdx.x;
  const int wave = tid >> 5, lane = tid & 31;
  const int hiL  = lane >> 4, l16 = lane & 15;
  const int rowbase = blockIdx.x * 16;
  const int row = tid >> 4, j = tid & 15;

  // ---- GEMM-O: acc = ctx @ Wo^T + bo ----
  const unsigned short* crow = Ctx + (size_t)(rowbase + l16) * DMODEL;
#pragma unroll
  for (int t = 0; t < 2; ++t) {
    const int n0 = (wave * 2 + t) * 16;
    const unsigned short* wrow = Wo + (size_t)(n0 + l16) * DMODEL;
    v8f c = {};
#pragma unroll
    for (int kt = 0; kt < 8; ++kt)
      c = wmma_bf16(load_frag(crow + kt * 32, hiL), load_frag(wrow + kt * 32, hiL), c);
    const float bv = bo[n0 + l16];
#pragma unroll
    for (int r = 0; r < 8; ++r) accS[r + 8 * hiL][n0 + l16] = c[r] + bv;
  }
  __syncthreads();

  // ---- LN1 (residual with x) -> x1S (f32) + stageS (bf16) ----
  {
    float y[16];
    float s = 0.0f;
#pragma unroll
    for (int i = 0; i < 16; ++i) {
      const int col = j * 16 + i;
      y[i] = accS[row][col] + X[(size_t)(rowbase + row) * DMODEL + col];
      s += y[i];
    }
#pragma unroll
    for (int o = 8; o; o >>= 1) s += __shfl_xor(s, o, 32);
    const float mu = s * (1.0f / 256.0f);
    float vs = 0.0f;
#pragma unroll
    for (int i = 0; i < 16; ++i) { const float d = y[i] - mu; vs += d * d; }
#pragma unroll
    for (int o = 8; o; o >>= 1) vs += __shfl_xor(vs, o, 32);
    const float rstd = rsqrtf(vs * (1.0f / 256.0f) + EPSLN);
#pragma unroll
    for (int i = 0; i < 16; ++i) {
      const int col = j * 16 + i;
      const float xv = (y[i] - mu) * rstd * g1[col] + be1[col];
      x1S[row][col] = xv;
      stageS[row][col] = f2bf(xv);
    }
  }
  __syncthreads();

  // ---- GEMM-1: acc = leaky(x1 @ W1^T + b1) ----
#pragma unroll
  for (int t = 0; t < 2; ++t) {
    const int n0 = (wave * 2 + t) * 16;
    const unsigned short* wrow = W1 + (size_t)(n0 + l16) * DMODEL;
    v8f c = {};
#pragma unroll
    for (int kt = 0; kt < 8; ++kt)
      c = wmma_bf16(load_frag(&stageS[l16][kt * 32], hiL),
                    load_frag(wrow + kt * 32, hiL), c);
    const float bv = b1[n0 + l16];
#pragma unroll
    for (int r = 0; r < 8; ++r) {
      float v = c[r] + bv;
      v = (v >= 0.0f) ? v : NEGSL * v;
      accS[r + 8 * hiL][n0 + l16] = v;
    }
  }
  __syncthreads();

  // ---- h -> bf16 stage ----
#pragma unroll
  for (int i = 0; i < 16; ++i) {
    const int col = j * 16 + i;
    stageS[row][col] = f2bf(accS[row][col]);
  }
  __syncthreads();

  // ---- GEMM-2: acc = h @ W2^T + b2 ----
#pragma unroll
  for (int t = 0; t < 2; ++t) {
    const int n0 = (wave * 2 + t) * 16;
    const unsigned short* wrow = W2 + (size_t)(n0 + l16) * DMODEL;
    v8f c = {};
#pragma unroll
    for (int kt = 0; kt < 8; ++kt)
      c = wmma_bf16(load_frag(&stageS[l16][kt * 32], hiL),
                    load_frag(wrow + kt * 32, hiL), c);
    const float bv = b2[n0 + l16];
#pragma unroll
    for (int r = 0; r < 8; ++r) accS[r + 8 * hiL][n0 + l16] = c[r] + bv;
  }
  __syncthreads();

  // ---- LN2 (residual with x1S) -> Out ----
  {
    float y[16];
    float s = 0.0f;
#pragma unroll
    for (int i = 0; i < 16; ++i) {
      const int col = j * 16 + i;
      y[i] = accS[row][col] + x1S[row][col];
      s += y[i];
    }
#pragma unroll
    for (int o = 8; o; o >>= 1) s += __shfl_xor(s, o, 32);
    const float mu = s * (1.0f / 256.0f);
    float vs = 0.0f;
#pragma unroll
    for (int i = 0; i < 16; ++i) { const float d = y[i] - mu; vs += d * d; }
#pragma unroll
    for (int o = 8; o; o >>= 1) vs += __shfl_xor(vs, o, 32);
    const float rstd = rsqrtf(vs * (1.0f / 256.0f) + EPSLN);
#pragma unroll
    for (int i = 0; i < 16; ++i) {
      const int col = j * 16 + i;
      Out[(size_t)(rowbase + row) * DMODEL + col] =
          (y[i] - mu) * rstd * g2[col] + be2[col];
    }
  }
}

// ---------------- host launch ----------------
extern "C" void kernel_launch(void* const* d_in, const int* in_sizes, int n_in,
                              void* d_out, int out_size, void* d_ws, size_t ws_size,
                              hipStream_t stream) {
  const float* x   = (const float*)d_in[0];
  const float* ep  = (const float*)d_in[1];
  const float* em  = (const float*)d_in[2];
  const float* Wq  = (const float*)d_in[3];
  const float* bq  = (const float*)d_in[4];
  const float* Wk  = (const float*)d_in[5];
  const float* bk  = (const float*)d_in[6];
  const float* Wv  = (const float*)d_in[7];
  const float* bv  = (const float*)d_in[8];
  const float* Wo  = (const float*)d_in[9];
  const float* bo  = (const float*)d_in[10];
  const float* W1  = (const float*)d_in[11];
  const float* b1  = (const float*)d_in[12];
  const float* W2  = (const float*)d_in[13];
  const float* b2  = (const float*)d_in[14];
  const float* g1  = (const float*)d_in[15];
  const float* be1 = (const float*)d_in[16];
  const float* g2  = (const float*)d_in[17];
  const float* be2 = (const float*)d_in[18];
  float* out = (float*)d_out;

  unsigned char* ws = (unsigned char*)d_ws;
  size_t off = 0;
  auto alloc = [&](size_t bytes) -> unsigned short* {
    unsigned short* p = (unsigned short*)(ws + off);
    off += (bytes + 255) & ~(size_t)255;
    return p;
  };
  unsigned short* xb  = alloc((size_t)NQTOT * DMODEL * 2);
  unsigned short* epb = alloc((size_t)MMEM  * DMODEL * 2);
  unsigned short* emb = alloc((size_t)MMEM  * DMODEL * 2);
  unsigned short* wqb = alloc((size_t)DMODEL * DMODEL * 2);
  unsigned short* wkb = alloc((size_t)DMODEL * DMODEL * 2);
  unsigned short* wvb = alloc((size_t)DMODEL * DMODEL * 2);
  unsigned short* wob = alloc((size_t)DMODEL * DMODEL * 2);
  unsigned short* w1b = alloc((size_t)DMODEL * DMODEL * 2);
  unsigned short* w2b = alloc((size_t)DMODEL * DMODEL * 2);
  unsigned short* qb  = alloc((size_t)NQTOT * DMODEL * 2);
  unsigned short* kb  = alloc((size_t)MMEM  * DMODEL * 2);
  unsigned short* vtb = alloc((size_t)DMODEL * MMEM * 2);   // V^T [256][2048]
  unsigned short* cxb = alloc((size_t)NQTOT * DMODEL * 2);

  const dim3 blk(256);

  // 0) bf16 conversions
  cvt_bf16_kernel<<<2048, blk, 0, stream>>>(x,  xb,  NQTOT * DMODEL);
  cvt_bf16_kernel<<<512,  blk, 0, stream>>>(ep, epb, MMEM * DMODEL);
  cvt_bf16_kernel<<<512,  blk, 0, stream>>>(em, emb, MMEM * DMODEL);
  cvt_bf16_kernel<<<64,   blk, 0, stream>>>(Wq, wqb, DMODEL * DMODEL);
  cvt_bf16_kernel<<<64,   blk, 0, stream>>>(Wk, wkb, DMODEL * DMODEL);
  cvt_bf16_kernel<<<64,   blk, 0, stream>>>(Wv, wvb, DMODEL * DMODEL);
  cvt_bf16_kernel<<<64,   blk, 0, stream>>>(Wo, wob, DMODEL * DMODEL);
  cvt_bf16_kernel<<<64,   blk, 0, stream>>>(W1, w1b, DMODEL * DMODEL);
  cvt_bf16_kernel<<<64,   blk, 0, stream>>>(W2, w2b, DMODEL * DMODEL);

  // 1) Q/K/V projections (V written transposed)
  proj_gemm_kernel<<<NQTOT / 16, blk, 0, stream>>>(xb,  wqb, bq, qb,  0, 0);
  proj_gemm_kernel<<<MMEM  / 16, blk, 0, stream>>>(epb, wkb, bk, kb,  0, 0);
  proj_gemm_kernel<<<MMEM  / 16, blk, 0, stream>>>(emb, wvb, bv, vtb, 1, MMEM);

  // 2) fused top-32 attention (LDS-resident scores)
  attn_kernel<<<dim3(NQTOT / 16, NHEAD), blk, SMEM_TOT, stream>>>(qb, kb, vtb, cxb);

  // 3) fused Wo-proj + LN1 + FFN + LN2
  epilogue_kernel<<<NQTOT / 16, blk, 0, stream>>>(cxb, x, wob, bo, g1, be1,
                                                  w1b, b1, w2b, b2, g2, be2, out);
}
